// SimpleNet_61332132987041
// MI455X (gfx1250) — matrix operand order repaired and checked by reading, not compiled, not run on hardware
//
#include <hip/hip_runtime.h>

typedef __attribute__((ext_vector_type(16))) _Float16 v16h;
typedef __attribute__((ext_vector_type(8)))  _Float16 v8h;
typedef __attribute__((ext_vector_type(8)))  float    v8f;
typedef __attribute__((ext_vector_type(4)))  float    v4f;

#define NMID     16
#define NLAYERS  18       // 0: W1 (K padded 2->32), 1: W2, 2..17: Ws[0..15]
#define WSTRIDE  40       // halves per weight row (bank-conflict-free b128 loads)
#define SSTRIDE  40       // halves per staging row
#define WAVES_PB 4
#define SUBT     2        // batch subtiles (16 rows each) per wave
#define ITERS_PW 8
#define BATCH    (1 << 20)
#define NPAIRS   (BATCH / 32)                       // 32768 (32 rows per wave-iter)
#define NBLOCKS  (NPAIRS / (WAVES_PB * ITERS_PW))   // 1024

struct LdsLayout {
    _Float16 wT[NLAYERS][32][WSTRIDE];   // W^T: [n][k]  (A-operand source)
    _Float16 w19T[16][WSTRIDE];          // W19^T padded to 16 output features
    float    bias[NLAYERS][32];
    float    bias19[16];
    float    maskL[NMID][32];
    _Float16 stage[WAVES_PB][SUBT][16][SSTRIDE]; // h tiles: [batch_row][feature]
};

__global__ __launch_bounds__(128)
void resmlp_wmma_kernel(const float* __restrict__ x,
                        const float* __restrict__ W1,  const float* __restrict__ b1,
                        const float* __restrict__ W2,  const float* __restrict__ b2,
                        const float* __restrict__ Ws,  const float* __restrict__ bs,
                        const float* __restrict__ masks,
                        const float* __restrict__ W19, const float* __restrict__ b19,
                        float* __restrict__ out)
{
    __shared__ LdsLayout lds;
    const int tid = threadIdx.x;

    // ---- one-time cooperative weight prep: transpose + cvt f32->f16 into LDS ----
    for (int i = tid; i < NLAYERS * 32 * 32; i += 128) {
        const int L = i >> 10;
        const int k = (i >> 5) & 31;   // input feature (K)
        const int n = i & 31;          // output feature (row of W^T)
        float v;
        if (L == 0)      v = (k < 2) ? W1[k * 32 + n] : 0.0f;
        else if (L == 1) v = W2[k * 32 + n];
        else             v = Ws[(L - 2) * 1024 + k * 32 + n];
        lds.wT[L][n][k] = (_Float16)v;
    }
    for (int i = tid; i < 16 * 32; i += 128) {
        const int n = i >> 5, k = i & 31;
        lds.w19T[n][k] = (_Float16)((n < 2) ? W19[k * 2 + n] : 0.0f);
    }
    for (int i = tid; i < NLAYERS * 32; i += 128) {
        const int L = i >> 5, d = i & 31;
        float v;
        if (L == 0)      v = b1[d];
        else if (L == 1) v = b2[d];
        else             v = bs[(L - 2) * 32 + d];
        lds.bias[L][d] = v;
    }
    if (tid < 16) lds.bias19[tid] = (tid < 2) ? b19[tid] : 0.0f;
    for (int i = tid; i < NMID * 32; i += 128)
        lds.maskL[i >> 5][i & 31] = masks[i];
    __syncthreads();

    const int wave = tid >> 5;
    const int lane = tid & 31;
    const int lo   = lane & 15;
    const int hi   = lane >> 4;
    _Float16* stg[SUBT] = { &lds.stage[wave][0][0][0], &lds.stage[wave][1][0][0] };

    // B-operand (activations, 32K x 16N): lane = batch col N=lo, halves[0..7]=K[hi*8..+7], halves[8..15]=K[16+hi*8..+7]
    auto loadBact = [&](const _Float16* s) -> v16h {
        v8h c0 = *(const v8h*)&s[lo * SSTRIDE + hi * 8];
        v8h c1 = *(const v8h*)&s[lo * SSTRIDE + 16 + hi * 8];
        return __builtin_shufflevector(c0, c1, 0,1,2,3,4,5,6,7,8,9,10,11,12,13,14,15);
    };
    // A-operand (weights, 16M x 32K): lane row M=lo (feature j*16+lo),
    // halves[0..7]=K[hi*8..+7], halves[8..15]=K[16+hi*8..+7]
    auto loadWA = [&](const _Float16* wbase, int j) -> v16h {
        const _Float16* r = wbase + (j * 16 + lo) * WSTRIDE;
        v8h c0 = *(const v8h*)(r + hi * 8);
        v8h c1 = *(const v8h*)(r + 16 + hi * 8);
        return __builtin_shufflevector(c0, c1, 0,1,2,3,4,5,6,7,8,9,10,11,12,13,14,15);
    };
    // D-layout (M=feature r+hi*8+16j, N=batch lo): 8 consecutive features per lane
    // -> single packed b128 store into row-major staging
    auto storeDh = [&](_Float16* s, const v8f& d, int j) {
        v8h hv;
        #pragma unroll
        for (int r = 0; r < 8; ++r) hv[r] = (_Float16)d[r];
        *(v8h*)&s[lo * SSTRIDE + j * 16 + hi * 8] = hv;
    };
    auto loadF8 = [&](const float* p) -> v8f {   // 8 consecutive floats, 2x b128
        v4f a = *(const v4f*)p;
        v4f b = *(const v4f*)(p + 4);
        return __builtin_shufflevector(a, b, 0,1,2,3,4,5,6,7);
    };
    auto relu8 = [&](v8f d) -> v8f {             // single v_med3 per element
        #pragma unroll
        for (int r = 0; r < 8; ++r)
            d[r] = __builtin_amdgcn_fmed3f(d[r], 0.0f, __builtin_inff());
        return d;
    };

    for (int it = 0; it < ITERS_PW; ++it) {
        const int p = blockIdx.x * WAVES_PB + wave + it * (NBLOCKS * WAVES_PB);
        const int rowBase = p * 32;

        // ---- stage x tiles, zero-padded K 2->32 ----
        #pragma unroll
        for (int s = 0; s < SUBT; ++s) {
            _Float16* sp = stg[s];
            v8h z = {};
            if (hi == 0) {
                float2 xr = *(const float2*)&x[(rowBase + s * 16 + lo) * 2];
                v8h c0 = {};
                c0[0] = (_Float16)xr.x;
                c0[1] = (_Float16)xr.y;
                *(v8h*)&sp[lo * SSTRIDE + 0] = c0;
                *(v8h*)&sp[lo * SSTRIDE + 8] = z;
            } else {
                *(v8h*)&sp[lo * SSTRIDE + 16] = z;
                *(v8h*)&sp[lo * SSTRIDE + 24] = z;
            }
        }

        v8f dP2[SUBT][2], dP[SUBT][2];

        // ---- fc1 ----
        {
            const _Float16* wb = &lds.wT[0][0][0];
            v16h wa0 = loadWA(wb, 0), wa1 = loadWA(wb, 1);
            v8f b0 = loadF8(&lds.bias[0][hi * 8]);
            v8f b1v = loadF8(&lds.bias[0][16 + hi * 8]);
            #pragma unroll
            for (int s = 0; s < SUBT; ++s) {
                v16h bact = loadBact(stg[s]);
                v8f d0 = __builtin_amdgcn_wmma_f32_16x16x32_f16(false, wa0, false, bact,
                                                                (short)0, b0, false, false);
                v8f d1 = __builtin_amdgcn_wmma_f32_16x16x32_f16(false, wa1, false, bact,
                                                                (short)0, b1v, false, false);
                d0 = relu8(d0); d1 = relu8(d1);
                dP2[s][0] = d0; dP2[s][1] = d1;
                storeDh(stg[s], d0, 0); storeDh(stg[s], d1, 1);
            }
        }

        // ---- fc2 ----
        {
            const _Float16* wb = &lds.wT[1][0][0];
            v16h wa0 = loadWA(wb, 0), wa1 = loadWA(wb, 1);
            v8f b0 = loadF8(&lds.bias[1][hi * 8]);
            v8f b1v = loadF8(&lds.bias[1][16 + hi * 8]);
            #pragma unroll
            for (int s = 0; s < SUBT; ++s) {
                v16h bact = loadBact(stg[s]);
                v8f d0 = __builtin_amdgcn_wmma_f32_16x16x32_f16(false, wa0, false, bact,
                                                                (short)0, b0, false, false);
                v8f d1 = __builtin_amdgcn_wmma_f32_16x16x32_f16(false, wa1, false, bact,
                                                                (short)0, b1v, false, false);
                d0 = relu8(d0); d1 = relu8(d1);
                dP[s][0] = d0; dP[s][1] = d1;
                storeDh(stg[s], d0, 0); storeDh(stg[s], d1, 1);
            }
        }

        // ---- fc3..fc18: h = relu(W h_prev + b + mask * h_prev2) ----
        for (int t = 0; t < NMID; ++t) {
            const _Float16* wb = &lds.wT[2 + t][0][0];
            v16h wa0 = loadWA(wb, 0), wa1 = loadWA(wb, 1);
            v8f b0 = loadF8(&lds.bias[2 + t][hi * 8]);
            v8f b1v = loadF8(&lds.bias[2 + t][16 + hi * 8]);
            v8f m0 = loadF8(&lds.maskL[t][hi * 8]);
            v8f m1 = loadF8(&lds.maskL[t][16 + hi * 8]);
            #pragma unroll
            for (int s = 0; s < SUBT; ++s) {
                v16h bact = loadBact(stg[s]);
                v8f c0, c1;
                #pragma unroll
                for (int r = 0; r < 8; ++r) {
                    c0[r] = fmaf(m0[r], dP2[s][0][r], b0[r]);
                    c1[r] = fmaf(m1[r], dP2[s][1][r], b1v[r]);
                }
                v8f d0 = __builtin_amdgcn_wmma_f32_16x16x32_f16(false, wa0, false, bact,
                                                                (short)0, c0, false, false);
                v8f d1 = __builtin_amdgcn_wmma_f32_16x16x32_f16(false, wa1, false, bact,
                                                                (short)0, c1, false, false);
                d0 = relu8(d0); d1 = relu8(d1);
                dP2[s][0] = dP[s][0]; dP2[s][1] = dP[s][1];
                dP[s][0] = d0;        dP[s][1] = d1;
                storeDh(stg[s], d0, 0); storeDh(stg[s], d1, 1);
            }
        }

        // ---- fc19: out = relu(W19p h18 + b19p); features 0,1 valid ----
        {
            v16h wa = loadWA(&lds.w19T[0][0], 0);
            v8f b19v = loadF8(&lds.bias19[hi * 8]);
            #pragma unroll
            for (int s = 0; s < SUBT; ++s) {
                v16h bact = loadBact(stg[s]);
                v8f d = __builtin_amdgcn_wmma_f32_16x16x32_f16(false, wa, false, bact,
                                                               (short)0, b19v, false, false);
                d = relu8(d);
                if (hi == 0) {   // lanes holding features 0..7; r=0,1 valid
                    float2 o;
                    o.x = d[0];
                    o.y = d[1];
                    *(float2*)&out[(rowBase + s * 16 + lo) * 2] = o;
                }
            }
        }
    }
}

extern "C" void kernel_launch(void* const* d_in, const int* in_sizes, int n_in,
                              void* d_out, int out_size, void* d_ws, size_t ws_size,
                              hipStream_t stream) {
    const float* x     = (const float*)d_in[0];
    const float* W1    = (const float*)d_in[1];
    const float* b1    = (const float*)d_in[2];
    const float* W2    = (const float*)d_in[3];
    const float* b2    = (const float*)d_in[4];
    const float* Ws    = (const float*)d_in[5];
    const float* bs    = (const float*)d_in[6];
    const float* masks = (const float*)d_in[7];
    const float* W19   = (const float*)d_in[8];
    const float* b19   = (const float*)d_in[9];
    float* out = (float*)d_out;

    hipLaunchKernelGGL(resmlp_wmma_kernel, dim3(NBLOCKS), dim3(128), 0, stream,
                       x, W1, b1, W2, b2, Ws, bs, masks, W19, b19, out);
}
